// MambaBlock_43705587204509
// MI455X (gfx1250) — compile-verified, matrix operand order checked by reading, not run on hardware
//
// MI455X (gfx1250) bidirectional GRU-Mamba block.
// - bf16 WMMA (v_wmma_f32_16x16x32_bf16, f32 accum) for all GEMMs.
// - Persistent grid-resident GRU scan: whh slice pinned in LDS (staged via
//   TDM tensor_load_to_lds with descriptor-side padding when available),
//   per-step gate inputs prefetched one step ahead via async global->LDS,
//   ds_add_f32 cross-wave reduction, one grid barrier per timestep.
#include <hip/hip_runtime.h>
#include <cstdint>
#include <cstddef>

typedef __attribute__((ext_vector_type(16))) __bf16 v16bf;
typedef __attribute__((ext_vector_type(8)))  __bf16 bf16x8;
typedef __attribute__((ext_vector_type(8)))  float  v8f;
typedef __attribute__((ext_vector_type(4)))  unsigned u32x4;
typedef __attribute__((ext_vector_type(8)))  int      i32x8;
typedef __attribute__((ext_vector_type(4)))  int      i32x4;

#define GLOBAL_AS __attribute__((address_space(1)))
#define LDS_AS    __attribute__((address_space(3)))

#if defined(__AMDGCN__) && __has_builtin(__builtin_amdgcn_global_load_async_to_lds_b128) && __has_builtin(__builtin_amdgcn_s_wait_asynccnt)
#define HAVE_ASYNC_LDS 1
#else
#define HAVE_ASYNC_LDS 0
#endif
#if defined(__AMDGCN__) && __has_builtin(__builtin_amdgcn_tensor_load_to_lds) && __has_builtin(__builtin_amdgcn_s_wait_tensorcnt)
#define HAVE_TDM 1
#else
#define HAVE_TDM 0
#endif

constexpr int BB = 8, LL = 2048, DD = 512, DI = 1024;
constexpr int BL = BB * LL;               // 16384 rows
constexpr int GBLK = 64, CHB = 16;        // scan: blocks/dir, channels/block
constexpr int ROWS = 3 * CHB;             // 48 whh rows per block
// LDS whh row stride: TDM pads 16B per 1024B chunk -> 1040 elems, 16B-aligned
// mid-row segment, 520-word bank stride.
constexpr int WPAD = DI + 16;

// ---------------------------------------------------------------- helpers
__device__ inline __bf16 f2bf(float f) {          // RNE f32 -> bf16
  unsigned u = __builtin_bit_cast(unsigned, f);
  u += 0x7FFFu + ((u >> 16) & 1u);
  unsigned short s = (unsigned short)(u >> 16);
  return __builtin_bit_cast(__bf16, s);
}
__device__ inline float sigm(float x) { return 1.0f / (1.0f + __expf(-x)); }

// 16x32 bf16 WMMA fragment (A layout; identical for (N,K)-major B/weights).
__device__ inline v16bf load_frag(const __bf16* p0, int ldk, int lane) {
  const __bf16* p = p0 + (size_t)(lane & 15) * ldk + (lane >> 4) * 8;
  bf16x8 lo = *(const bf16x8*)p;
  bf16x8 hi = *(const bf16x8*)(p + 16);
  v16bf r;
#pragma unroll
  for (int i = 0; i < 8; ++i) { r[i] = lo[i]; r[8 + i] = hi[i]; }
  return r;
}
#define WMMA_BF16(a, b, c) \
  __builtin_amdgcn_wmma_f32_16x16x32_bf16(false, (a), false, (b), (short)0, (c), false, false)

// ---------------------------------------------------------------- kernels
__global__ void zero_i32_kernel(int* p, int n) {
  for (int i = blockIdx.x * 256 + threadIdx.x; i < n; i += gridDim.x * 256) p[i] = 0;
}

__global__ void cvt_bf16_kernel(const float* __restrict__ s, __bf16* __restrict__ d, int n) {
  for (int i = blockIdx.x * 256 + threadIdx.x; i < n; i += gridDim.x * 256) d[i] = f2bf(s[i]);
}

// LayerNorm(x) under both (w1,b1) and (w2,b2); second copy written L-flipped.
__global__ void __launch_bounds__(128)
ln_dual_kernel(const float* __restrict__ x,
               const float* __restrict__ w1, const float* __restrict__ b1,
               const float* __restrict__ w2, const float* __restrict__ b2,
               __bf16* __restrict__ xf, __bf16* __restrict__ xb) {
  __shared__ float red[128];
  int row = blockIdx.x, tid = threadIdx.x;
  const float* xr = x + (size_t)row * DD;
  float v[4], s = 0.f;
#pragma unroll
  for (int j = 0; j < 4; ++j) { v[j] = xr[tid + j * 128]; s += v[j]; }
  red[tid] = s; __syncthreads();
  for (int st = 64; st > 0; st >>= 1) { if (tid < st) red[tid] += red[tid + st]; __syncthreads(); }
  float mu = red[0] * (1.0f / DD);
  __syncthreads();
  float q = 0.f;
#pragma unroll
  for (int j = 0; j < 4; ++j) { float d = v[j] - mu; q += d * d; }
  red[tid] = q; __syncthreads();
  for (int st = 64; st > 0; st >>= 1) { if (tid < st) red[tid] += red[tid + st]; __syncthreads(); }
  float rs = rsqrtf(red[0] * (1.0f / DD) + 1e-5f);
  int b = row / LL, l = row % LL;
  size_t fo = (size_t)row * DD;
  size_t bo = ((size_t)b * LL + (LL - 1 - l)) * DD;
#pragma unroll
  for (int j = 0; j < 4; ++j) {
    int i = tid + j * 128;
    float nv = (v[j] - mu) * rs;
    xf[fo + i] = f2bf(nv * w1[i] + b1[i]);
    xb[bo + i] = f2bf(nv * w2[i] + b2[i]);
  }
}

// Generic C = A(M,K)bf16 * W(N,K)bf16^T, f32 accumulate via WMMA.
// mode 0: f32 store. mode 1: bf16 store (colOff, optional L-flip of rows).
// mode 2: f32 store + bias[n] + resid[row].
__global__ void __launch_bounds__(256)
gemm_bf16_wmma(const __bf16* __restrict__ A, const __bf16* __restrict__ W,
               int M, int N, int K,
               float* outF, __bf16* outH, int outLd, int colOff,
               const float* __restrict__ bias, const float* __restrict__ resid,
               int flipL, int mode) {
  int lane = threadIdx.x & 31, w = threadIdx.x >> 5;
  int wm = w >> 2, wn = w & 3;                       // 2(M) x 4(N) waves
  int m0 = blockIdx.y * 64 + wm * 32;
  int n0 = blockIdx.x * 128 + wn * 32;
  v8f c00{}, c01{}, c10{}, c11{};
  const __bf16* a0p = A + (size_t)m0 * K;
  const __bf16* a1p = A + (size_t)(m0 + 16) * K;
  const __bf16* b0p = W + (size_t)n0 * K;
  const __bf16* b1p = W + (size_t)(n0 + 16) * K;
  for (int k = 0; k < K; k += 32) {
    __builtin_prefetch(a0p + k + 256, 0, 1);
    __builtin_prefetch(b0p + k + 256, 0, 1);
    v16bf a0 = load_frag(a0p + k, K, lane);
    v16bf a1 = load_frag(a1p + k, K, lane);
    v16bf b0 = load_frag(b0p + k, K, lane);
    v16bf b1 = load_frag(b1p + k, K, lane);
    c00 = WMMA_BF16(a0, b0, c00);
    c01 = WMMA_BF16(a0, b1, c01);
    c10 = WMMA_BF16(a1, b0, c10);
    c11 = WMMA_BF16(a1, b1, c11);
  }
  int nl = lane & 15, mset = lane >> 4;
#pragma unroll
  for (int t = 0; t < 4; ++t) {
    v8f c = (t == 0) ? c00 : (t == 1) ? c01 : (t == 2) ? c10 : c11;
    int tm = m0 + (t >> 1) * 16;
    int tn = n0 + (t & 1) * 16 + nl;
#pragma unroll
    for (int v = 0; v < 8; ++v) {
      int m = tm + v + 8 * mset;
      int row = m;
      if (flipL) { int b = m / LL, l = m % LL; row = b * LL + (LL - 1 - l); }
      size_t o = (size_t)row * outLd + colOff + tn;
      float val = c[v];
      if (mode == 0)      outF[o] = val;
      else if (mode == 1) outH[o] = f2bf(val);
      else                outF[o] = val + bias[tn] + resid[(size_t)m * outLd + tn];
    }
  }
}

// Depthwise causal conv(K=4) + SiLU; reads xc = xz[...,:DI] f32, writes bf16.
__global__ void __launch_bounds__(256)
conv_silu_kernel(const float* __restrict__ xz, const float* __restrict__ cw,
                 const float* __restrict__ cb, __bf16* __restrict__ y) {
  size_t idx = (size_t)blockIdx.x * 256 + threadIdx.x;   // over BL*DI
  int c = (int)(idx % DI);
  size_t bl = idx / DI;
  int l = (int)(bl % LL);
  float acc = cb[c];
#pragma unroll
  for (int t = 0; t < 4; ++t) {
    int ls = l - 3 + t;
    if (ls >= 0) acc += cw[c * 4 + t] * xz[(bl + (size_t)(ls - l)) * (2 * DI) + c];
  }
  y[idx] = f2bf(acc * sigm(acc));
}

struct DirPtrs {
  const float* gx; const float* xz; const __bf16* whh;
  const float* bih; const float* bhh;
  __bf16* hp0; __bf16* hp1; float* hm; __bf16* u;
};

// Async prefetch of next-step gate inputs (gx r/z/n slices + xz z-gate) into
// LDS double buffer: 128 lanes x 16B = 2KB per step, hidden under the WMMAs.
__device__ inline void issue_gate_prefetch(const DirPtrs& P, int c0, int lnext,
                                           float* pf) {
#if HAVE_ASYNC_LDS
  int tid = threadIdx.x;
  if (tid < 128) {
    int seg = tid >> 2, quad = tid & 3;
    const float* src;
    float* dst;
    if (seg < 24) {
      int gate = seg >> 3, i = seg & 7;
      src = P.gx + ((size_t)i * LL + lnext) * (3 * DI) + gate * DI + c0 + quad * 4;
      dst = pf + seg * 16 + quad * 4;
    } else {
      int i = seg - 24;
      src = P.xz + ((size_t)i * LL + lnext) * (2 * DI) + DI + c0 + quad * 4;
      dst = pf + 384 + i * 16 + quad * 4;
    }
    __builtin_amdgcn_global_load_async_to_lds_b128(
        (GLOBAL_AS i32x4*)src, (LDS_AS i32x4*)dst, 0, 0);
  }
#else
  (void)P; (void)c0; (void)lnext; (void)pf;
#endif
}

// Persistent GRU scan, both directions concurrently (blocks [0,64)=fwd,
// [64,128)=bwd). whh 48x1024 slice pinned in LDS for 2048 steps; one
// grid-wide sense-reversing barrier per step; h ping-pongs in bf16.
__global__ void __launch_bounds__(256)
gru_scan_kernel(DirPtrs fw, DirPtrs bw, int* bar, int nblocks) {
  extern __shared__ __align__(16) char smem[];
  __bf16* whh_s = (__bf16*)smem;                               // 48*1040 bf16
  float* gh_s = (float*)(smem + (size_t)ROWS * WPAD * 2);      // 768 f32
  float* pf_s = (float*)(smem + (size_t)ROWS * WPAD * 2 + 3072); // 2*512 f32
  int tid = threadIdx.x;
  int dir = blockIdx.x / GBLK, g = blockIdx.x % GBLK;
  DirPtrs P = dir ? bw : fw;
  int c0 = g * CHB;

  // ---- stage whh slice (3 gate blocks of 16x1024 bf16) into LDS ----
#if HAVE_TDM
  if (tid == 0) {
    unsigned lbase = (unsigned)(size_t)(LDS_AS char*)smem;
    for (int t = 0; t < 3; ++t) {
      unsigned long long ga =
          (unsigned long long)(size_t)(P.whh + (size_t)(t * DI + c0) * DI);
      u32x4 g0;
      g0[0] = 1u;                                     // count=1, user mode
      g0[1] = lbase + (unsigned)(t * 16 * WPAD * 2);  // lds_addr (bytes)
      g0[2] = (unsigned)ga;                           // global_addr[31:0]
      g0[3] = (unsigned)(ga >> 32) | 0x80000000u;     // addr[56:32] | type=2
      i32x8 g1;
      // data_size=2B, pad_enable, pad_interval=256 DW, pad_amount=4 DW
      g1[0] = (1 << 16) | (1 << 20) | (7 << 22) | (3 << 25);
      g1[1] = (int)(1024u << 16);   // tensor_dim0 = 1024 (low16 @ bits63:48)
      g1[2] = (int)(16u << 16);     // tensor_dim1 = 16  (low16 @ bits95:80)
      g1[3] = (int)(1024u << 16);   // tile_dim0 = 1024  (bits127:112)
      g1[4] = 16;                   // tile_dim1 = 16, tile_dim2 = 0
      g1[5] = 1024;                 // tensor_dim0_stride = 1024
      g1[6] = 0;
      g1[7] = 0;
      i32x4 gz4 = {0, 0, 0, 0};     // 2D tensor: groups 2/3 unused
      i32x8 gz8 = {0, 0, 0, 0, 0, 0, 0, 0};
      __builtin_amdgcn_tensor_load_to_lds(g0, g1, gz4, gz4, gz8, 0);
    }
    __builtin_amdgcn_s_wait_tensorcnt(0);
  }
#else
  for (int t = tid; t < ROWS * (DI / 8); t += 256) {   // mirror TDM pad layout
    int rr = t / (DI / 8), ck = t % (DI / 8);
    int grow = (rr / CHB) * DI + c0 + (rr % CHB);
    int dst = rr * WPAD + ck * 8 + ((ck >= 64) ? 8 : 0);
    *(bf16x8*)(whh_s + dst) = *(const bf16x8*)(P.whh + (size_t)grow * DI + ck * 8);
  }
#endif
  // prologue: prefetch gate inputs for step 0 into buffer 0
  issue_gate_prefetch(P, c0, 0, pf_s);
  __syncthreads();

  int lane = tid & 31, w = tid >> 5;
  int nl = lane & 15, mset = lane >> 4;
  for (int l = 0; l < LL; ++l) {
    __bf16* hin  = (l & 1) ? P.hp1 : P.hp0;
    __bf16* hout = (l & 1) ? P.hp0 : P.hp1;
    // kick next step's gate-input copy; wait only for the previous one
    issue_gate_prefetch(P, c0, l + 1, pf_s + (((l + 1) & 1) << 9));
#if HAVE_ASYNC_LDS
    if (tid < 128) __builtin_amdgcn_s_wait_asynccnt(1);
#endif
    for (int t = tid; t < 3 * 256; t += 256) gh_s[t] = 0.f;
    __syncthreads();
    v8f cg0{}, cg1{}, cg2{};                              // gh = h @ whh_sl^T
#pragma unroll
    for (int kk = 0; kk < 4; ++kk) {
      int k = w * 128 + kk * 32;                          // K split over waves
      int ka = k + ((k >= 512) ? 8 : 0);                  // skip mid-row pad
      v16bf a  = load_frag(hin + k, DI, lane);
      v16bf b0 = load_frag(whh_s + (size_t)(0 * 16) * WPAD + ka, WPAD, lane);
      v16bf b1 = load_frag(whh_s + (size_t)(1 * 16) * WPAD + ka, WPAD, lane);
      v16bf b2 = load_frag(whh_s + (size_t)(2 * 16) * WPAD + ka, WPAD, lane);
      cg0 = WMMA_BF16(a, b0, cg0);
      cg1 = WMMA_BF16(a, b1, cg1);
      cg2 = WMMA_BF16(a, b2, cg2);
    }
#pragma unroll
    for (int v = 0; v < 8; ++v) {                         // ds_add_f32 reduce
      int m = v + 8 * mset;
      atomicAdd(&gh_s[0 * 256 + m * 16 + nl], cg0[v]);
      atomicAdd(&gh_s[1 * 256 + m * 16 + nl], cg1[v]);
      atomicAdd(&gh_s[2 * 256 + m * 16 + nl], cg2[v]);
    }
    __syncthreads();
    if (tid < 128) {                                      // gates + h update
      int cl = tid >> 3, i = tid & 7, c = c0 + cl;
      float ghr = gh_s[0 * 256 + i * 16 + cl];
      float ghz = gh_s[1 * 256 + i * 16 + cl];
      float ghn = gh_s[2 * 256 + i * 16 + cl];
#if HAVE_ASYNC_LDS
      const float* pf = pf_s + ((l & 1) << 9);
      float gxr = pf[(0 * 8 + i) * 16 + cl];
      float gxz = pf[(1 * 8 + i) * 16 + cl];
      float gxn = pf[(2 * 8 + i) * 16 + cl];
      float zg  = pf[384 + i * 16 + cl];
#else
      const float* gxp = P.gx + ((size_t)i * LL + l) * (3 * DI);
      float gxr = gxp[c];
      float gxz = gxp[DI + c];
      float gxn = gxp[2 * DI + c];
      float zg  = P.xz[((size_t)i * LL + l) * (2 * DI) + DI + c];
#endif
      float r = sigm(gxr + P.bih[c] + ghr + P.bhh[c]);
      float z = sigm(gxz + P.bih[DI + c] + ghz + P.bhh[DI + c]);
      float n = tanhf(gxn + P.bih[2 * DI + c] + r * (ghn + P.bhh[2 * DI + c]));
      float hold = P.hm[i * DI + c];
      float hnew = (1.f - z) * n + z * hold;
      P.hm[i * DI + c] = hnew;
      hout[i * DI + c] = f2bf(hnew);
      P.u[((size_t)i * LL + l) * DI + c] = f2bf(hnew * (zg * sigm(zg)));
    }
    __threadfence();
    __syncthreads();
    if (tid == 0) {                                       // grid barrier
      int gen = __hip_atomic_load(bar + 1, __ATOMIC_RELAXED, __HIP_MEMORY_SCOPE_AGENT);
      if (__hip_atomic_fetch_add(bar, 1, __ATOMIC_ACQ_REL, __HIP_MEMORY_SCOPE_AGENT) == nblocks - 1) {
        __hip_atomic_store(bar, 0, __ATOMIC_RELAXED, __HIP_MEMORY_SCOPE_AGENT);
        __hip_atomic_fetch_add(bar + 1, 1, __ATOMIC_RELEASE, __HIP_MEMORY_SCOPE_AGENT);
      } else {
        while (__hip_atomic_load(bar + 1, __ATOMIC_ACQUIRE, __HIP_MEMORY_SCOPE_AGENT) == gen) {
          __builtin_amdgcn_s_sleep(1);
        }
      }
    }
    __syncthreads();
  }
}

// ---------------------------------------------------------------- host
extern "C" void kernel_launch(void* const* d_in, const int* in_sizes, int n_in,
                              void* d_out, int out_size, void* d_ws, size_t ws_size,
                              hipStream_t stream) {
  (void)in_sizes; (void)n_in; (void)out_size; (void)ws_size;
  const float* x   = (const float*)d_in[0];
  const float* n1w = (const float*)d_in[1];
  const float* n1b = (const float*)d_in[2];
  const float* n2w = (const float*)d_in[3];
  const float* n2b = (const float*)d_in[4];
  const float* fusw = (const float*)d_in[5];
  const float* fusb = (const float*)d_in[6];
  const float* inp_f32[2], *convw[2], *convb[2], *wih_f32[2], *whh_f32[2];
  const float* bih[2], *bhh[2], *outp_f32[2];
  for (int d = 0; d < 2; ++d) {
    int o = 7 + d * 8;
    inp_f32[d]  = (const float*)d_in[o + 0];
    convw[d]    = (const float*)d_in[o + 1];
    convb[d]    = (const float*)d_in[o + 2];
    wih_f32[d]  = (const float*)d_in[o + 3];
    whh_f32[d]  = (const float*)d_in[o + 4];
    bih[d]      = (const float*)d_in[o + 5];
    bhh[d]      = (const float*)d_in[o + 6];
    outp_f32[d] = (const float*)d_in[o + 7];
  }
  char* ws = (char*)d_ws;
  size_t off = 0;
  auto alloc = [&](size_t bytes) { size_t o = off; off += (bytes + 255) & ~(size_t)255; return o; };
  size_t o_winp[2], o_wih[2], o_whh[2], o_wout[2], o_xln[2], o_xz[2], o_y[2], o_gx[2], o_u[2];
  for (int d = 0; d < 2; ++d) {
    o_winp[d] = alloc((size_t)2 * DI * DD * 2);
    o_wih[d]  = alloc((size_t)3 * DI * DI * 2);
    o_whh[d]  = alloc((size_t)3 * DI * DI * 2);
    o_wout[d] = alloc((size_t)DD * DI * 2);
  }
  size_t o_wfus = alloc((size_t)DD * 2 * DD * 2);
  for (int d = 0; d < 2; ++d) {
    o_xln[d] = alloc((size_t)BL * DD * 2);
    o_xz[d]  = alloc((size_t)BL * 2 * DI * 4);
    o_y[d]   = alloc((size_t)BL * DI * 2);
    o_gx[d]  = alloc((size_t)BL * 3 * DI * 4);
    o_u[d]   = alloc((size_t)BL * DI * 2);
  }
  size_t o_comb = alloc((size_t)BL * 2 * DD * 2);
  size_t o_hp   = alloc((size_t)4 * 16 * DI * 2);   // 4 ping-pong h (bf16)
  size_t o_hm   = alloc((size_t)2 * BB * DI * 4);   // 2 f32 master h
  size_t o_bar  = alloc(256);
  // zero h state + barrier (contiguous region)
  zero_i32_kernel<<<64, 256, 0, stream>>>((int*)(ws + o_hp),
      (int)((o_bar + 256 - o_hp) / 4));
  // weights -> bf16
  auto cvt = [&](const float* s, size_t o, int n) {
    cvt_bf16_kernel<<<256, 256, 0, stream>>>(s, (__bf16*)(ws + o), n);
  };
  for (int d = 0; d < 2; ++d) {
    cvt(inp_f32[d],  o_winp[d], 2 * DI * DD);
    cvt(wih_f32[d],  o_wih[d],  3 * DI * DI);
    cvt(whh_f32[d],  o_whh[d],  3 * DI * DI);
    cvt(outp_f32[d], o_wout[d], DD * DI);
  }
  cvt(fusw, o_wfus, DD * 2 * DD);
  // dual layernorm (bwd copy pre-flipped)
  ln_dual_kernel<<<BL, 128, 0, stream>>>(x, n1w, n1b, n2w, n2b,
      (__bf16*)(ws + o_xln[0]), (__bf16*)(ws + o_xln[1]));
  // in_proj: xz = xln @ in_proj^T   (M=16384, N=2048, K=512)
  for (int d = 0; d < 2; ++d)
    gemm_bf16_wmma<<<dim3(2 * DI / 128, BL / 64), 256, 0, stream>>>(
        (const __bf16*)(ws + o_xln[d]), (const __bf16*)(ws + o_winp[d]),
        BL, 2 * DI, DD, (float*)(ws + o_xz[d]), nullptr, 2 * DI, 0,
        nullptr, nullptr, 0, 0);
  // depthwise conv + SiLU
  for (int d = 0; d < 2; ++d)
    conv_silu_kernel<<<BL * DI / 256, 256, 0, stream>>>(
        (const float*)(ws + o_xz[d]), convw[d], convb[d], (__bf16*)(ws + o_y[d]));
  // gx = y @ wih^T   (N=3072, K=1024)
  for (int d = 0; d < 2; ++d)
    gemm_bf16_wmma<<<dim3(3 * DI / 128, BL / 64), 256, 0, stream>>>(
        (const __bf16*)(ws + o_y[d]), (const __bf16*)(ws + o_wih[d]),
        BL, 3 * DI, DI, (float*)(ws + o_gx[d]), nullptr, 3 * DI, 0,
        nullptr, nullptr, 0, 0);
  // persistent GRU scan, both directions
  DirPtrs dp[2];
  for (int d = 0; d < 2; ++d) {
    dp[d].gx  = (const float*)(ws + o_gx[d]);
    dp[d].xz  = (const float*)(ws + o_xz[d]);
    dp[d].whh = (const __bf16*)(ws + o_whh[d]);
    dp[d].bih = bih[d]; dp[d].bhh = bhh[d];
    dp[d].hp0 = (__bf16*)(ws + o_hp + (size_t)(2 * d) * 16 * DI * 2);
    dp[d].hp1 = (__bf16*)(ws + o_hp + (size_t)(2 * d + 1) * 16 * DI * 2);
    dp[d].hm  = (float*)(ws + o_hm + (size_t)d * BB * DI * 4);
    dp[d].u   = (__bf16*)(ws + o_u[d]);
  }
  size_t scan_lds = (size_t)ROWS * WPAD * 2 + 3 * 256 * 4 + 2 * 512 * 4;
  gru_scan_kernel<<<2 * GBLK, 256, scan_lds, stream>>>(dp[0], dp[1],
      (int*)(ws + o_bar), 2 * GBLK);
  // out_proj -> comb (bwd half un-flipped)   (N=512, K=1024)
  gemm_bf16_wmma<<<dim3(DD / 128, BL / 64), 256, 0, stream>>>(
      (const __bf16*)(ws + o_u[0]), (const __bf16*)(ws + o_wout[0]),
      BL, DD, DI, nullptr, (__bf16*)(ws + o_comb), 2 * DD, 0,
      nullptr, nullptr, 0, 1);
  gemm_bf16_wmma<<<dim3(DD / 128, BL / 64), 256, 0, stream>>>(
      (const __bf16*)(ws + o_u[1]), (const __bf16*)(ws + o_wout[1]),
      BL, DD, DI, nullptr, (__bf16*)(ws + o_comb), 2 * DD, DD,
      nullptr, nullptr, 1, 1);
  // fusion: out = comb @ fusion_w^T + fusion_b + x   (N=512, K=1024)
  gemm_bf16_wmma<<<dim3(DD / 128, BL / 64), 256, 0, stream>>>(
      (const __bf16*)(ws + o_comb), (const __bf16*)(ws + o_wfus),
      BL, DD, 2 * DD, (float*)d_out, nullptr, DD, 0,
      fusb, x, 0, 2);
}
